// XASNet_PNA_12996571037720
// MI455X (gfx1250) — matrix operand order repaired and checked by reading, not compiled, not run on hardware
//
#include <hip/hip_runtime.h>
#include <hip/hip_bf16.h>
#include <cstdint>
#include <cstddef>

typedef _Float16 h16;
typedef __attribute__((ext_vector_type(16))) _Float16 v16h;
typedef __attribute__((ext_vector_type(8)))  float    v8f;

#define EPS_BN  1e-5f
#define EPS_STD 1e-5f

// ---------- order-preserving float <-> uint mapping for atomic min/max ----------
__device__ __forceinline__ unsigned ordKey(float f) {
  unsigned u = __float_as_uint(f);
  return (u & 0x80000000u) ? ~u : (u | 0x80000000u);
}
__device__ __forceinline__ float ordVal(unsigned u) {
  return (u & 0x80000000u) ? __uint_as_float(u & 0x7fffffffu) : __uint_as_float(~u);
}

// ---------- WMMA fragment loaders (wave32, 16-bit A 16x32 / B 32x16 layouts) ----------
// A tile LDS layout: As[row][k] (row-major, 40-half stride -> 16B aligned rows).
// Per-lane A data is two contiguous 16B runs -> ds_load_b128 x2.
__device__ __forceinline__ v16h load_a_frag(h16 As[][40], int rowBase, int lane) {
  const int hi = lane >> 4, ml = lane & 15;
  v16h a;
#pragma unroll
  for (int e = 0; e < 16; ++e) {
    const int k = hi * 16 + e;
    a[e] = As[rowBase + ml][k];
  }
  return a;
}
// B strip LDS layout is TRANSPOSED: BsT[col][k]. A lane's 16 B values
// (K = hi*16 .. hi*16+15 at fixed col) are 32 contiguous bytes -> ds_load_b128 x2.
__device__ __forceinline__ v16h load_b_frag(h16 BsT[][40], int colOff, int lane) {
  const int hi = lane >> 4, ml = lane & 15;
  v16h b;
#pragma unroll
  for (int e = 0; e < 16; ++e) b[e] = BsT[colOff + ml][hi * 16 + e];
  return b;
}

// =====================================================================
// Edge message GEMM:  m = [x_dst | x_src | (edge_attr@edge_w+edge_b)] @ pre_w + pre_b
// fused with PNA aggregation (atomic sum / sumsq / min / max over dst).
// Block: 128 threads (4 wave32), 64 edges x 64 cols; K staged in 32-strips.
// =====================================================================
template <int F>
__global__ __launch_bounds__(128)
void edge_msg_gemm(const float* __restrict__ h,
                   const int* __restrict__ src, const int* __restrict__ dst,
                   const float* __restrict__ eattr,
                   const float* __restrict__ edge_w, const float* __restrict__ edge_b,
                   const h16* __restrict__ preW, const float* __restrict__ pre_b,
                   float* __restrict__ sumB, float* __restrict__ sqB,
                   unsigned* __restrict__ mnk, unsigned* __restrict__ mxk,
                   int nE) {
  constexpr int K   = 3 * F;
  constexpr int KT  = (K + 31) / 32;
  constexpr int NTL = (F >= 64) ? 4 : ((F + 15) / 16);
  const int rowBase = blockIdx.x * 64;
  const int colBase = blockIdx.y * 64;
  const int tid = threadIdx.x, lane = tid & 31, wv = tid >> 5;
  const int hi = lane >> 4, ml = lane & 15;

  __shared__ h16 As[64][40];
  __shared__ h16 BsT[64][40];
  __shared__ int   sDst[64], sSrc[64];
  __shared__ float sEA[64][4];

  if (tid < 64) {
    int e = rowBase + tid;
    int ok = (e < nE);
    sDst[tid]   = ok ? dst[e] : -1;
    sSrc[tid]   = ok ? src[e] : 0;
    sEA[tid][0] = ok ? eattr[e * 3 + 0] : 0.f;
    sEA[tid][1] = ok ? eattr[e * 3 + 1] : 0.f;
    sEA[tid][2] = ok ? eattr[e * 3 + 2] : 0.f;
  }

  v8f acc[NTL] = {};

  for (int kt = 0; kt < KT; ++kt) {
    // prefetch next weight strip (one 128B line per row of the strip)
    if (tid < 32 && (kt + 1) < KT) {
      int k = (kt + 1) * 32 + tid;
      if (k < K) __builtin_prefetch(&preW[(size_t)k * F + colBase], 0, 0);
    }
    __syncthreads();
    // stage A tile (64 edges x 32 K) as f16
    for (int i = tid; i < 64 * 32; i += 128) {
      int r = i >> 5, kk = i & 31, k = kt * 32 + kk;
      float v = 0.f;
      if (sDst[r] >= 0 && k < K) {
        if (k < F)            v = h[(size_t)sDst[r] * F + k];
        else if (k < 2 * F)   v = h[(size_t)sSrc[r] * F + (k - F)];
        else {
          int j = k - 2 * F;
          v = edge_b[j] + sEA[r][0] * edge_w[j]
                        + sEA[r][1] * edge_w[F + j]
                        + sEA[r][2] * edge_w[2 * F + j];
        }
      }
      As[r][kk] = (h16)v;
    }
    // stage B strip (32 K x 64 cols) of pre_w, transposed into LDS
    for (int i = tid; i < 32 * 64; i += 128) {
      int r = i >> 6, c = i & 63;
      int k = kt * 32 + r, col = colBase + c;
      h16 v = (h16)0.f;
      if (k < K && col < F) v = preW[(size_t)k * F + col];
      BsT[c][r] = v;
    }
    __syncthreads();

    v16h a = load_a_frag(As, wv * 16, lane);
#pragma unroll
    for (int nt = 0; nt < NTL; ++nt) {
      v16h b = load_b_frag(BsT, nt * 16, lane);
      acc[nt] = __builtin_amdgcn_wmma_f32_16x16x32_f16(false, a, false, b,
                                                       (short)0, acc[nt], false, false);
    }
  }

  // fused aggregation epilogue straight from accumulators
#pragma unroll
  for (int nt = 0; nt < NTL; ++nt) {
    int col = colBase + nt * 16 + ml;
    if (col < F) {
      float pb = pre_b[col];
#pragma unroll
      for (int r = 0; r < 8; ++r) {
        int lr = wv * 16 + hi * 8 + r;
        int d  = sDst[lr];
        if (d >= 0) {
          float v = acc[nt][r] + pb;
          size_t o = (size_t)d * F + col;
          atomicAdd(&sumB[o], v);
          atomicAdd(&sqB[o], v * v);
          atomicMin(&mnk[o], ordKey(v));
          atomicMax(&mxk[o], ordKey(v));
        }
      }
    }
  }
}

// =====================================================================
// Post GEMM: out = [x | agg | agg*amp | agg*att] @ post_w + post_b
// agg = [mean, min, max, std] computed inline from the atomic aggregates.
// =====================================================================
template <int FIN, int FOUT>
__global__ __launch_bounds__(128)
void post_gemm(const float* __restrict__ h,
               const float* __restrict__ sumB, const float* __restrict__ sqB,
               const unsigned* __restrict__ mnk, const unsigned* __restrict__ mxk,
               const int* __restrict__ cnt,
               const float* __restrict__ amp, const float* __restrict__ att,
               const h16* __restrict__ postW, const float* __restrict__ post_b,
               float* __restrict__ out, int nN) {
  constexpr int K  = 13 * FIN;
  constexpr int KT = (K + 31) / 32;
  const int rowBase = blockIdx.x * 64;
  const int colBase = blockIdx.y * 64;
  const int tid = threadIdx.x, lane = tid & 31, wv = tid >> 5;
  const int hi = lane >> 4, ml = lane & 15;

  __shared__ h16 As[64][40];
  __shared__ h16 BsT[64][40];
  __shared__ float sCnt[64], sAmp[64], sAtt[64];
  __shared__ int sOk[64];

  if (tid < 64) {
    int n = rowBase + tid;
    int ok = (n < nN);
    sOk[tid]  = ok;
    sCnt[tid] = ok ? (float)cnt[n] : 0.f;
    sAmp[tid] = ok ? amp[n] : 0.f;
    sAtt[tid] = ok ? att[n] : 0.f;
  }

  v8f acc[4] = {};

  for (int kt = 0; kt < KT; ++kt) {
    if (tid < 32 && (kt + 1) < KT) {
      int k = (kt + 1) * 32 + tid;
      if (k < K) __builtin_prefetch(&postW[(size_t)k * FOUT + colBase], 0, 0);
    }
    __syncthreads();
    for (int i = tid; i < 64 * 32; i += 128) {
      int r = i >> 5, kk = i & 31, k = kt * 32 + kk;
      int n = rowBase + r;
      float v = 0.f;
      if (sOk[r] && k < K) {
        if (k < FIN) {
          v = h[(size_t)n * FIN + k];
        } else {
          int kp = k - FIN;
          int g = kp / FIN;
          int c = kp - g * FIN;
          size_t o = (size_t)n * FIN + c;
          float cn = sCnt[r], cc = fmaxf(cn, 1.f);
          int ai = g & 3;
          float val;
          if (ai == 0)      val = sumB[o] / cc;
          else if (ai == 1) val = (cn > 0.f) ? ordVal(mnk[o]) : 0.f;
          else if (ai == 2) val = (cn > 0.f) ? ordVal(mxk[o]) : 0.f;
          else {
            float mu = sumB[o] / cc;
            float vr = sqB[o] / cc - mu * mu;
            val = sqrtf(fmaxf(vr, 0.f) + EPS_STD);
          }
          float s = (g < 4) ? 1.f : ((g < 8) ? sAmp[r] : sAtt[r]);
          v = val * s;
        }
      }
      As[r][kk] = (h16)v;
    }
    for (int i = tid; i < 32 * 64; i += 128) {
      int r = i >> 6, c = i & 63;
      int k = kt * 32 + r, col = colBase + c;
      h16 v = (h16)0.f;
      if (k < K) v = postW[(size_t)k * FOUT + col];
      BsT[c][r] = v;
    }
    __syncthreads();

    v16h a = load_a_frag(As, wv * 16, lane);
#pragma unroll
    for (int nt = 0; nt < 4; ++nt) {
      v16h b = load_b_frag(BsT, nt * 16, lane);
      acc[nt] = __builtin_amdgcn_wmma_f32_16x16x32_f16(false, a, false, b,
                                                       (short)0, acc[nt], false, false);
    }
  }

#pragma unroll
  for (int nt = 0; nt < 4; ++nt) {
    int col = colBase + nt * 16 + ml;
    float pb = post_b[col];
#pragma unroll
    for (int r = 0; r < 8; ++r) {
      int lr = wv * 16 + hi * 8 + r;
      int n = rowBase + lr;
      if (sOk[lr]) out[(size_t)n * FOUT + col] = acc[nt][r] + pb;
    }
  }
}

// =====================================================================
// Lin GEMM + bias + BatchNorm(eval) + ReLU:  h_next = relu(bn(tmp @ lin_w + lin_b))
// =====================================================================
template <int FOUT>
__global__ __launch_bounds__(128)
void lin_bn_gemm(const float* __restrict__ inA,
                 const h16* __restrict__ linW, const float* __restrict__ lin_b,
                 const float* __restrict__ bng, const float* __restrict__ bnb,
                 const float* __restrict__ bnm, const float* __restrict__ bnv,
                 float* __restrict__ hout, int nN) {
  constexpr int K  = FOUT;
  constexpr int KT = (K + 31) / 32;
  const int rowBase = blockIdx.x * 64;
  const int colBase = blockIdx.y * 64;
  const int tid = threadIdx.x, lane = tid & 31, wv = tid >> 5;
  const int hi = lane >> 4, ml = lane & 15;

  __shared__ h16 As[64][40];
  __shared__ h16 BsT[64][40];

  v8f acc[4] = {};

  for (int kt = 0; kt < KT; ++kt) {
    if (tid < 32 && (kt + 1) < KT) {
      int k = (kt + 1) * 32 + tid;
      if (k < K) __builtin_prefetch(&linW[(size_t)k * FOUT + colBase], 0, 0);
    }
    __syncthreads();
    for (int i = tid; i < 64 * 32; i += 128) {
      int r = i >> 5, kk = i & 31, k = kt * 32 + kk;
      int n = rowBase + r;
      float v = (n < nN && k < K) ? inA[(size_t)n * K + k] : 0.f;
      As[r][kk] = (h16)v;
    }
    for (int i = tid; i < 32 * 64; i += 128) {
      int r = i >> 6, c = i & 63;
      int k = kt * 32 + r, col = colBase + c;
      h16 v = (h16)0.f;
      if (k < K) v = linW[(size_t)k * FOUT + col];
      BsT[c][r] = v;
    }
    __syncthreads();

    v16h a = load_a_frag(As, wv * 16, lane);
#pragma unroll
    for (int nt = 0; nt < 4; ++nt) {
      v16h b = load_b_frag(BsT, nt * 16, lane);
      acc[nt] = __builtin_amdgcn_wmma_f32_16x16x32_f16(false, a, false, b,
                                                       (short)0, acc[nt], false, false);
    }
  }

#pragma unroll
  for (int nt = 0; nt < 4; ++nt) {
    int col = colBase + nt * 16 + ml;
    float lb = lin_b[col];
    float sc = bng[col] * rsqrtf(bnv[col] + EPS_BN);
    float mm = bnm[col], bb = bnb[col];
#pragma unroll
    for (int r = 0; r < 8; ++r) {
      int lr = wv * 16 + hi * 8 + r;
      int n = rowBase + lr;
      if (n < nN) {
        float v = acc[nt][r] + lb;
        v = (v - mm) * sc + bb;
        hout[(size_t)n * FOUT + col] = fmaxf(v, 0.f);
      }
    }
  }
}

// ---------------------------- small helper kernels ----------------------------
__global__ void fill_u32(unsigned* p, unsigned v, int n) {
  int i = blockIdx.x * blockDim.x + threadIdx.x;
  if (i < n) p[i] = v;
}
__global__ void cvt_f16(const float* __restrict__ in, h16* __restrict__ out, int n) {
  int i = blockIdx.x * blockDim.x + threadIdx.x;
  if (i < n) out[i] = (h16)in[i];
}
__global__ void degree_kernel(const int* __restrict__ dst, int* cnt, int E) {
  int i = blockIdx.x * blockDim.x + threadIdx.x;
  if (i < E) atomicAdd(&cnt[dst[i]], 1);
}
__global__ void avg_log_kernel(const float* __restrict__ dh, int L, float* out) {
  __shared__ float s0[256], s1[256];
  float a = 0.f, b = 0.f;
  for (int i = threadIdx.x; i < L; i += 256) {
    float hh = dh[i];
    a += logf((float)i + 1.f) * hh;
    b += hh;
  }
  s0[threadIdx.x] = a; s1[threadIdx.x] = b;
  __syncthreads();
  for (int o = 128; o > 0; o >>= 1) {
    if (threadIdx.x < o) { s0[threadIdx.x] += s0[threadIdx.x + o]; s1[threadIdx.x] += s1[threadIdx.x + o]; }
    __syncthreads();
  }
  if (threadIdx.x == 0) out[0] = s0[0] / s1[0];
}
__global__ void scalers_kernel(const int* __restrict__ cnt, const float* __restrict__ avgLog,
                               float* amp, float* att, int n) {
  int i = blockIdx.x * blockDim.x + threadIdx.x;
  if (i < n) {
    float d = fmaxf((float)cnt[i], 1.f);
    float l = logf(d + 1.f);
    float al = avgLog[0];
    amp[i] = l / al;
    att[i] = al / l;
  }
}
__global__ void pool_sum(const float* __restrict__ h, const int* __restrict__ bseg,
                         float* gsum, int N) {
  int i = blockIdx.x * blockDim.x + threadIdx.x;
  if (i < N * 256) {
    int n = i >> 8, c = i & 255;
    atomicAdd(&gsum[(size_t)bseg[n] * 256 + c], h[i]);
  }
}
__global__ void pool_cnt(const int* __restrict__ bseg, int* gcnt, int N) {
  int i = blockIdx.x * blockDim.x + threadIdx.x;
  if (i < N) atomicAdd(&gcnt[bseg[i]], 1);
}
__global__ void mlp_kernel(const float* __restrict__ gsum, const int* __restrict__ gcnt,
                           const float* __restrict__ w, const float* __restrict__ b,
                           float* out, int G, int T) {
  int i = blockIdx.x * blockDim.x + threadIdx.x;
  if (i < G * T) {
    int g = i / T, t = i - g * T;
    float inv = 1.f / fmaxf((float)gcnt[g], 1.f);
    float acc = b[t];
    for (int k = 0; k < 256; ++k) acc += gsum[(size_t)g * 256 + k] * inv * w[(size_t)k * T + t];
    out[i] = (acc >= 0.f) ? acc : 0.1f * acc;
  }
}

// ---------------------------- per-layer host helper ----------------------------
template <int F, int FO>
static void run_layer(const float* hin, float* hout,
                      const int* src, const int* dstI, const float* eattr,
                      const float* const* prm,
                      const h16* preW16, const h16* postW16, const h16* linW16,
                      float* sumB, float* sqB, unsigned* mnk, unsigned* mxk,
                      const int* cnt, const float* amp, const float* att,
                      float* tmp, int N, int E, hipStream_t stream) {
  int nf = N * F;
  int fb = (nf + 255) / 256;
  fill_u32<<<fb, 256, 0, stream>>>((unsigned*)sumB, 0u, nf);
  fill_u32<<<fb, 256, 0, stream>>>((unsigned*)sqB, 0u, nf);
  fill_u32<<<fb, 256, 0, stream>>>(mnk, 0xFFFFFFFFu, nf);
  fill_u32<<<fb, 256, 0, stream>>>(mxk, 0u, nf);

  dim3 ge((E + 63) / 64, (F + 63) / 64);
  edge_msg_gemm<F><<<ge, 128, 0, stream>>>(hin, src, dstI, eattr,
                                           prm[0], prm[1], preW16, prm[3],
                                           sumB, sqB, mnk, mxk, E);
  dim3 gp((N + 63) / 64, FO / 64);
  post_gemm<F, FO><<<gp, 128, 0, stream>>>(hin, sumB, sqB, mnk, mxk, cnt, amp, att,
                                           postW16, prm[5], tmp, N);
  lin_bn_gemm<FO><<<gp, 128, 0, stream>>>(tmp, linW16, prm[7],
                                          prm[8], prm[9], prm[10], prm[11], hout, N);
}

// ---------------------------- entry point ----------------------------
extern "C" void kernel_launch(void* const* d_in, const int* in_sizes, int n_in,
                              void* d_out, int out_size, void* d_ws, size_t ws_size,
                              hipStream_t stream) {
  (void)n_in; (void)ws_size;
  const float* x     = (const float*)d_in[0];
  const int*   ei    = (const int*)d_in[1];
  const float* eattr = (const float*)d_in[2];
  const int*   bseg  = (const int*)d_in[3];
  const float* dh    = (const float*)d_in[4];

  const int N = in_sizes[0] / 5;
  const int E = in_sizes[1] / 2;
  const int L = in_sizes[4];
  const int T = 100;
  const int G = out_size / T;

  const int* src  = ei;
  const int* dstI = ei + E;

  const float* prm[3][12];
  for (int c = 0; c < 3; ++c)
    for (int j = 0; j < 12; ++j)
      prm[c][j] = (const float*)d_in[5 + 12 * c + j];
  const float* mlp_w = (const float*)d_in[41];
  const float* mlp_b = (const float*)d_in[42];

  // ---- workspace carve (bump allocator, 256B aligned) ----
  char* wsp = (char*)d_ws;
  auto take = [&](size_t bytes) -> char* {
    char* r = wsp;
    wsp += (bytes + 255) & ~(size_t)255;
    return r;
  };
  float*    avgLog = (float*)take(4);
  int*      cnt    = (int*)take((size_t)N * 4);
  float*    amp    = (float*)take((size_t)N * 4);
  float*    att    = (float*)take((size_t)N * 4);
  float*    sumB   = (float*)take((size_t)N * 128 * 4);
  float*    sqB    = (float*)take((size_t)N * 128 * 4);
  unsigned* mnk    = (unsigned*)take((size_t)N * 128 * 4);
  unsigned* mxk    = (unsigned*)take((size_t)N * 128 * 4);
  float*    tmp    = (float*)take((size_t)N * 256 * 4);
  float*    h1     = (float*)take((size_t)N * 64 * 4);
  float*    h2     = (float*)take((size_t)N * 128 * 4);
  float*    h3     = (float*)take((size_t)N * 256 * 4);
  float*    gsum   = (float*)take((size_t)G * 256 * 4);
  int*      gcnt   = (int*)take((size_t)G * 4);
  const int preSz[3]  = {15 * 5, 192 * 64, 384 * 128};
  const int postSz[3] = {65 * 64, 832 * 128, 1664 * 256};
  const int linSz[3]  = {64 * 64, 128 * 128, 256 * 256};
  h16* preW16[3]; h16* postW16[3]; h16* linW16[3];
  for (int c = 0; c < 3; ++c) {
    preW16[c]  = (h16*)take((size_t)preSz[c] * 2);
    postW16[c] = (h16*)take((size_t)postSz[c] * 2);
    linW16[c]  = (h16*)take((size_t)linSz[c] * 2);
  }

  // ---- one-time per call: degree, scalers, f16 weight conversion ----
  fill_u32<<<(N + 255) / 256, 256, 0, stream>>>((unsigned*)cnt, 0u, N);
  degree_kernel<<<(E + 255) / 256, 256, 0, stream>>>(dstI, cnt, E);
  avg_log_kernel<<<1, 256, 0, stream>>>(dh, L, avgLog);
  scalers_kernel<<<(N + 255) / 256, 256, 0, stream>>>(cnt, avgLog, amp, att, N);
  for (int c = 0; c < 3; ++c) {
    cvt_f16<<<(preSz[c] + 255) / 256, 256, 0, stream>>>(prm[c][2], preW16[c], preSz[c]);
    cvt_f16<<<(postSz[c] + 255) / 256, 256, 0, stream>>>(prm[c][4], postW16[c], postSz[c]);
    cvt_f16<<<(linSz[c] + 255) / 256, 256, 0, stream>>>(prm[c][6], linW16[c], linSz[c]);
  }

  // ---- three PNA conv layers ----
  run_layer<5, 64>(x, h1, src, dstI, eattr, prm[0], preW16[0], postW16[0], linW16[0],
                   sumB, sqB, mnk, mxk, cnt, amp, att, tmp, N, E, stream);
  run_layer<64, 128>(h1, h2, src, dstI, eattr, prm[1], preW16[1], postW16[1], linW16[1],
                     sumB, sqB, mnk, mxk, cnt, amp, att, tmp, N, E, stream);
  run_layer<128, 256>(h2, h3, src, dstI, eattr, prm[2], preW16[2], postW16[2], linW16[2],
                      sumB, sqB, mnk, mxk, cnt, amp, att, tmp, N, E, stream);

  // ---- global mean pool + MLP + LeakyReLU ----
  fill_u32<<<(G * 256 + 255) / 256, 256, 0, stream>>>((unsigned*)gsum, 0u, G * 256);
  fill_u32<<<(G + 255) / 256, 256, 0, stream>>>((unsigned*)gcnt, 0u, G);
  pool_sum<<<((size_t)N * 256 + 255) / 256, 256, 0, stream>>>(h3, bseg, gsum, N);
  pool_cnt<<<(N + 255) / 256, 256, 0, stream>>>(bseg, gcnt, N);
  mlp_kernel<<<(G * T + 127) / 128, 128, 0, stream>>>(gsum, gcnt, mlp_w, mlp_b,
                                                      (float*)d_out, G, T);
}
